// LocalGraph_32633161515662
// MI455X (gfx1250) — matrix operand order repaired and checked by reading, not compiled, not run on hardware
//
#include <hip/hip_runtime.h>

// MI455X / gfx1250 LocalGraph block.
// All three 50176x256x256 GEMMs + the per-frame 49x49x256 similarity GEMM run on
// v_wmma_f32_16x16x32_bf16 (bf16 in, f32 acc), fed by packed b128 fragment loads
// from pre-converted bf16 buffers. Graph aggregation is frame-local (LDS atomics).

typedef __bf16 bf16;
typedef __attribute__((ext_vector_type(16))) __bf16 v16bf;
typedef __attribute__((ext_vector_type(8)))  __bf16 v8bf;
typedef __attribute__((ext_vector_type(8)))  float  v8f;

#define C_DIM 256
#define N_POS 49
#define B_IMG 1024
#define M_TOT (B_IMG * N_POS)   // 50176 rows = 3136 M-tiles of 16 = 784 blocks of 4
#define EPSV  1e-5f

// Build a 16-element bf16 fragment from two contiguous 8-element (16B) loads.
// 'second' = element offset of the upper half (A and B: +8).
__device__ __forceinline__ v16bf load_frag16(const bf16* p, int second) {
    v8bf lo = *(const v8bf*)p;
    v8bf hi = *(const v8bf*)(p + second);
    return __builtin_shufflevector(lo, hi, 0, 1, 2, 3, 4, 5, 6, 7,
                                   8, 9, 10, 11, 12, 13, 14, 15);
}

__device__ __forceinline__ v8f wmma_bf16(v16bf a, v16bf b, v8f c) {
    return __builtin_amdgcn_wmma_f32_16x16x32_bf16(false, a, false, b,
                                                   (short)0, c, false, false);
}

// ---------------- BN affine precompute -------------------------------
// out = (dot + bias - mean)*g*rsqrt(v+eps) + beta  ==  dot*A + B
__global__ void bn_prep_kernel(const float* __restrict__ b1, const float* __restrict__ g1,
                               const float* __restrict__ be1, const float* __restrict__ m1,
                               const float* __restrict__ v1,
                               const float* __restrict__ b2, const float* __restrict__ g2,
                               const float* __restrict__ be2, const float* __restrict__ m2,
                               const float* __restrict__ v2,
                               float* __restrict__ aA1, float* __restrict__ aB1,
                               float* __restrict__ aA2, float* __restrict__ aB2) {
    int c = threadIdx.x;
    if (c < C_DIM) {
        float s1 = g1[c] * rsqrtf(v1[c] + EPSV);
        aA1[c] = s1;
        aB1[c] = (b1[c] - m1[c]) * s1 + be1[c];
        float s2 = g2[c] * rsqrtf(v2[c] + EPSV);
        aA2[c] = s2;
        aB2[c] = (b2[c] - m2[c]) * s2 + be2[c];
    }
}

// ---------------- one-time f32 -> bf16 conversions -------------------
__global__ void cvt_kernel(const float* __restrict__ in, bf16* __restrict__ out, int n) {
    for (int i = blockIdx.x * blockDim.x + threadIdx.x; i < n; i += gridDim.x * blockDim.x)
        out[i] = (bf16)in[i];
}

// x [B, C, 7, 7] f32  ->  xrow [B*49, 256] bf16  (transpose + convert, once)
__global__ void transpose_x_kernel(const float* __restrict__ x, bf16* __restrict__ xr) {
    int b = blockIdx.x;
    const float* xb = x + (size_t)b * C_DIM * N_POS;
    bf16* xo = xr + (size_t)b * N_POS * C_DIM;
    for (int e = threadIdx.x; e < C_DIM * N_POS; e += blockDim.x) {
        int c = e / N_POS, p = e - c * N_POS;
        xo[p * C_DIM + c] = (bf16)xb[e];
    }
}

// ---------------- unified WMMA GEMM: [M,256]bf16 x [O,256]^T bf16 ----
// Each wave: 16(M) x 64(N) strip, A fragment reused across 4 WMMAs/K-step.
// MODE 0: affine epilogue -> bf16 row-major (down-proj + BN1)
// MODE 1: raw f32 row-major (gcn x W)
// MODE 2: affine epilogue -> f32 NCHW scatter (up-proj + BN2)
template <int MODE>
__global__ void gemm_kernel(const bf16* __restrict__ A, const bf16* __restrict__ W,
                            const float* __restrict__ aA, const float* __restrict__ aB,
                            void* __restrict__ outp) {
    int lane  = threadIdx.x & 31;
    int wave  = threadIdx.x >> 5;
    int mtile = blockIdx.x * 4 + wave;
    int nbase = blockIdx.y * 4;
    int r15   = lane & 15;
    int kbA   = (lane & 16) ? 16 : 0;   // A-fragment sub-K offset (contiguous k = 16h + s)
    int kbB   = (lane & 16) ? 16 : 0;   // B-fragment sub-K offset
    const bf16* arow = A + (size_t)(mtile * 16 + r15) * C_DIM + kbA;
    const bf16* wcol = W + (size_t)(nbase * 16 + r15) * C_DIM + kbB;
    __builtin_prefetch(arow + (size_t)64 * C_DIM, 0, 1);   // next M-panel
    v8f acc[4] = {{}, {}, {}, {}};
#pragma unroll
    for (int k0 = 0; k0 < C_DIM; k0 += 32) {
        v16bf a  = load_frag16(arow + k0, 8);
        v16bf b0 = load_frag16(wcol + k0, 8);
        v16bf b1 = load_frag16(wcol + 16 * C_DIM + k0, 8);
        v16bf b2 = load_frag16(wcol + 32 * C_DIM + k0, 8);
        v16bf b3 = load_frag16(wcol + 48 * C_DIM + k0, 8);
        acc[0] = wmma_bf16(a, b0, acc[0]);
        acc[1] = wmma_bf16(a, b1, acc[1]);
        acc[2] = wmma_bf16(a, b2, acc[2]);
        acc[3] = wmma_bf16(a, b3, acc[3]);
    }
    int mbase = mtile * 16 + ((lane >> 4) << 3);   // D layout: VGPR r <-> row r+8*(lane>=16)
#pragma unroll
    for (int nt = 0; nt < 4; ++nt) {
        int ocol = (nbase + nt) * 16 + r15;
        if constexpr (MODE == 0) {
            float sc = aA[ocol], sh = aB[ocol];
            bf16* o = (bf16*)outp;
#pragma unroll
            for (int rr = 0; rr < 8; ++rr)
                o[(size_t)(mbase + rr) * C_DIM + ocol] = (bf16)(acc[nt][rr] * sc + sh);
        } else if constexpr (MODE == 1) {
            float* o = (float*)outp;
#pragma unroll
            for (int rr = 0; rr < 8; ++rr)
                o[(size_t)(mbase + rr) * C_DIM + ocol] = acc[nt][rr];
        } else {
            float sc = aA[ocol], sh = aB[ocol];
            float* o = (float*)outp;
#pragma unroll
            for (int rr = 0; rr < 8; ++rr) {
                int mo = mbase + rr;
                int bo = mo / N_POS;
                int po = mo - bo * N_POS;
                o[((size_t)bo * C_DIM + ocol) * N_POS + po] = acc[nt][rr] * sc + sh;
            }
        }
    }
}

// ---------------- similarity: per frame  -(xf.xf^T + rel_pos) --------
// 49 padded to 64: rows/cols >= 49 clamp to 48 (finite garbage, stores masked).
__global__ void sim_kernel(const bf16* __restrict__ xf,
                           const float* __restrict__ rel_pos,   // [49,49]
                           float* __restrict__ sim) {
    int lane = threadIdx.x & 31;
    int b  = blockIdx.x;
    int it = blockIdx.y;
    int jt = blockIdx.z;
    int r15 = lane & 15;
    int i = min(it * 16 + r15, N_POS - 1);
    int j = min(jt * 16 + r15, N_POS - 1);
    int kbA = (lane & 16) ? 16 : 0;
    int kbB = (lane & 16) ? 16 : 0;
    const bf16* base = xf + (size_t)b * N_POS * C_DIM;
    const bf16* ra = base + (size_t)i * C_DIM + kbA;
    const bf16* rb = base + (size_t)j * C_DIM + kbB;
    v8f acc = {};
#pragma unroll
    for (int k0 = 0; k0 < C_DIM; k0 += 32) {
        v16bf av = load_frag16(ra + k0, 8);
        v16bf bv = load_frag16(rb + k0, 8);
        acc = wmma_bf16(av, bv, acc);
    }
    int jo    = jt * 16 + r15;
    int ibase = it * 16 + ((lane >> 4) << 3);
#pragma unroll
    for (int rr = 0; rr < 8; ++rr) {
        int io = ibase + rr;
        if (io < N_POS && jo < N_POS)
            sim[(size_t)b * (N_POS * N_POS) + io * N_POS + jo] =
                -(acc[rr] + rel_pos[io * N_POS + jo]);
    }
}

// ---------------- per-frame normalize + threshold + top-4 ------------
__global__ void topk_kernel(const float* __restrict__ sim, int* __restrict__ nbr) {
    __shared__ float red[64];
    int b = blockIdx.x, t = threadIdx.x;
    const float* s = sim + (size_t)b * (N_POS * N_POS);
    float ss = 0.f;
    for (int e = t; e < N_POS * N_POS; e += 64) { float v = s[e]; ss += v * v; }
    red[t] = ss;
    __syncthreads();
    for (int off = 32; off > 0; off >>= 1) {
        if (t < off) red[t] += red[t + off];
        __syncthreads();
    }
    float inv = rsqrtf(red[0] > 0.f ? red[0] : 1.f);
    if (t < N_POS) {
        float vals[N_POS];
        for (int j = 0; j < N_POS; ++j) {
            float v = s[t * N_POS + j] * inv;
            vals[j] = (v < 0.05f) ? 100.0f : v;        // jnp.where(s<0.05, 100, s)
        }
        for (int k = 0; k < 4; ++k) {                  // lax.top_k: ties -> lowest idx
            float best = -1e30f; int bi = 0;
            for (int j = 0; j < N_POS; ++j)
                if (vals[j] > best) { best = vals[j]; bi = j; }
            nbr[((size_t)b * N_POS + t) * 4 + k] = bi;
            vals[bi] = -1e30f;
        }
    }
}

// ---------------- frame-local GCN aggregation ------------------------
// deg = 5 + indeg(nbr); out[i] = dinv[i]*( x~[i] + sum_k x~[nbr[i,k]]
//        + sum_{j,k: nbr[j,k]==i} x~[j] ) + gcn_b,  x~[s] = xw[s]*dinv[s]
__global__ void aggregate_kernel(const float* __restrict__ xw,
                                 const int* __restrict__ nbr,
                                 const float* __restrict__ gcn_b,
                                 bf16* __restrict__ y) {
    extern __shared__ float smem[];
    float* wrow = smem;                        // 49*256
    float* acc  = wrow + N_POS * C_DIM;        // 49*256
    float* deg  = acc  + N_POS * C_DIM;        // 49
    float* dinv = deg  + N_POS;                // 49
    int*   nb   = (int*)(dinv + N_POS);        // 196
    int b = blockIdx.x, t = threadIdx.x;
    if (t < N_POS) deg[t] = 5.0f;              // K as-src + self loop
    if (t < N_POS * 4) nb[t] = nbr[(size_t)b * N_POS * 4 + t];
    __syncthreads();
    if (t < N_POS * 4) atomicAdd(&deg[nb[t]], 1.0f);
    __syncthreads();
    if (t < N_POS) dinv[t] = rsqrtf(deg[t]);
    __syncthreads();
    const float* xb = xw + (size_t)b * N_POS * C_DIM;
    for (int e = t; e < N_POS * C_DIM; e += blockDim.x) {
        int i = e >> 8;
        float v = xb[e] * dinv[i];
        wrow[e] = v;
        acc[e]  = v;                           // self loop
    }
    __syncthreads();
    for (int idx = t; idx < N_POS * 4 * C_DIM; idx += blockDim.x) {
        int e = idx >> 8;                      // edge 0..195
        int c = idx & 255;
        int i = e >> 2;                        // source row
        int d = nb[e];                         // neighbor
        atomicAdd(&acc[d * C_DIM + c], wrow[i * C_DIM + c]);   // forward
        atomicAdd(&acc[i * C_DIM + c], wrow[d * C_DIM + c]);   // reverse
    }
    __syncthreads();
    bf16* yb = y + (size_t)b * N_POS * C_DIM;
    for (int e = t; e < N_POS * C_DIM; e += blockDim.x) {
        int i = e >> 8, c = e & 255;
        yb[e] = (bf16)(acc[e] * dinv[i] + gcn_b[c]);
    }
}

extern "C" void kernel_launch(void* const* d_in, const int* in_sizes, int n_in,
                              void* d_out, int out_size, void* d_ws, size_t ws_size,
                              hipStream_t stream) {
    const float* x      = (const float*)d_in[0];
    const float* down_w = (const float*)d_in[1];
    const float* down_b = (const float*)d_in[2];
    const float* bn1_g  = (const float*)d_in[3];
    const float* bn1_b  = (const float*)d_in[4];
    const float* bn1_m  = (const float*)d_in[5];
    const float* bn1_v  = (const float*)d_in[6];
    const float* rel_pos= (const float*)d_in[7];
    const float* gcn_w  = (const float*)d_in[8];
    const float* gcn_b  = (const float*)d_in[9];
    const float* up_w   = (const float*)d_in[10];
    const float* up_b   = (const float*)d_in[11];
    const float* bn2_g  = (const float*)d_in[12];
    const float* bn2_b  = (const float*)d_in[13];
    const float* bn2_m  = (const float*)d_in[14];
    const float* bn2_v  = (const float*)d_in[15];
    float* out = (float*)d_out;
    (void)ws_size; (void)in_sizes; (void)n_in; (void)out_size;

    // ---- workspace carve-up (~104 MB) ----
    char* ws = (char*)d_ws;
    float* aA1 = (float*)(ws + 0);
    float* aB1 = (float*)(ws + 1024);
    float* aA2 = (float*)(ws + 2048);
    float* aB2 = (float*)(ws + 3072);
    size_t off = 4096;
    bf16* wd_bf = (bf16*)(ws + off); off += (size_t)C_DIM * C_DIM * 2;   // 128 KB each
    bf16* wg_bf = (bf16*)(ws + off); off += (size_t)C_DIM * C_DIM * 2;
    bf16* wu_bf = (bf16*)(ws + off); off += (size_t)C_DIM * C_DIM * 2;
    bf16* xrow_bf = (bf16*)(ws + off);                 // 25.7 MB, input rows bf16
    bf16* y_bf    = xrow_bf;                           // reused after down-proj GEMM
    off += (size_t)M_TOT * C_DIM * 2;
    bf16* xf_bf = (bf16*)(ws + off);                   // 25.7 MB, down-projected nodes
    off += (size_t)M_TOT * C_DIM * 2;
    int* nbr = (int*)(ws + off);                       // 0.8 MB
    off += (size_t)B_IMG * N_POS * 4 * sizeof(int);
    float* sim = (float*)(ws + off);                   // 9.8 MB, dead after topk
    float* xw  = (float*)(ws + off);                   // 51.4 MB, reuses sim region

    bn_prep_kernel<<<1, 256, 0, stream>>>(down_b, bn1_g, bn1_b, bn1_m, bn1_v,
                                          up_b,   bn2_g, bn2_b, bn2_m, bn2_v,
                                          aA1, aB1, aA2, aB2);

    cvt_kernel<<<256, 256, 0, stream>>>(down_w, wd_bf, C_DIM * C_DIM);
    cvt_kernel<<<256, 256, 0, stream>>>(gcn_w,  wg_bf, C_DIM * C_DIM);
    cvt_kernel<<<256, 256, 0, stream>>>(up_w,   wu_bf, C_DIM * C_DIM);
    transpose_x_kernel<<<B_IMG, 256, 0, stream>>>(x, xrow_bf);

    dim3 ggrid(M_TOT / 64, 4);                         // (784, 4); 4 waves x 16x64 strip
    gemm_kernel<0><<<ggrid, 128, 0, stream>>>(xrow_bf, wd_bf, aA1, aB1, xf_bf);

    sim_kernel<<<dim3(B_IMG, 4, 4), 32, 0, stream>>>(xf_bf, rel_pos, sim);

    topk_kernel<<<B_IMG, 64, 0, stream>>>(sim, nbr);

    gemm_kernel<1><<<ggrid, 128, 0, stream>>>(xf_bf, wg_bf, aA1, aB1, xw);

    size_t lds_bytes = (size_t)(2 * N_POS * C_DIM) * sizeof(float)
                     + (size_t)(2 * N_POS) * sizeof(float)
                     + (size_t)(N_POS * 4) * sizeof(int);
    aggregate_kernel<<<B_IMG, 256, lds_bytes, stream>>>(xw, nbr, gcn_b, y_bf);

    gemm_kernel<2><<<ggrid, 128, 0, stream>>>(y_bf, wu_bf, aA2, aB2, out);
}
